// MultiHeadedAttention_11158325035146
// MI455X (gfx1250) — matrix operand order repaired and checked by reading, not compiled, not run on hardware
//
#include <hip/hip_runtime.h>
#include <hip/hip_bf16.h>

#define B_  4
#define S_  2048
#define D_  1024
#define H_  16
#define DH_ 64
#define BS_ (B_*S_)

typedef __attribute__((ext_vector_type(16))) __bf16 v16bf;
typedef __attribute__((ext_vector_type(8)))  float  v8f;
typedef __attribute__((ext_vector_type(4)))  unsigned v4u;
typedef __attribute__((ext_vector_type(8)))  unsigned v8u;

union Frag {
    v16bf v;
    uint4 q[2];
    unsigned short s[16];
};

__device__ __forceinline__ unsigned short f2bf(float f) {
    unsigned int u = __float_as_uint(f);
    u += 0x7FFFu + ((u >> 16) & 1u);        // round-to-nearest-even
    return (unsigned short)(u >> 16);
}

__device__ __forceinline__ v8f v8f_zero() {
    v8f z = {0.f,0.f,0.f,0.f,0.f,0.f,0.f,0.f};
    return z;
}

__device__ __forceinline__ v8f wmma_bf16(v16bf a, v16bf b, v8f c) {
    // D = A(16x32 bf16) * B(32x16 bf16) + C(16x16 f32)
    return __builtin_amdgcn_wmma_f32_16x16x32_bf16(
        false, a, false, b, (short)0, c, false, false);
}

// reduce across the 16 lanes of each half-wave (N dimension of a C/D tile)
__device__ __forceinline__ float rmax16(float x) {
    x = fmaxf(x, __shfl_xor(x, 1, 32));
    x = fmaxf(x, __shfl_xor(x, 2, 32));
    x = fmaxf(x, __shfl_xor(x, 4, 32));
    x = fmaxf(x, __shfl_xor(x, 8, 32));
    return x;
}
__device__ __forceinline__ float rsum16(float x) {
    x += __shfl_xor(x, 1, 32);
    x += __shfl_xor(x, 2, 32);
    x += __shfl_xor(x, 4, 32);
    x += __shfl_xor(x, 8, 32);
    return x;
}

// ---- Tensor Data Mover: 2D tile load, D# built per ISA ch.8 (groups 0+1) ----
__device__ __forceinline__ void tdm_load_2d(unsigned lds_addr, const void* gptr,
                                            unsigned tensor_d0, unsigned tensor_d1,
                                            unsigned tile_d0, unsigned tile_d1,
                                            unsigned long long stride0)
{
    unsigned long long ga = (unsigned long long)(uintptr_t)gptr;
    v4u g0;
    g0[0] = 1u;                                             // count=1, user D#
    g0[1] = lds_addr;                                       // LDS byte address
    g0[2] = (unsigned)ga;                                   // global addr [31:0]
    g0[3] = (unsigned)((ga >> 32) & 0x01FFFFFFu) | (2u << 30); // [56:32] | type=2
    v8u g1;
    g1[0] = (1u << 16);                                     // data_size=1 (2B), wg_mask=0
    g1[1] = (tensor_d0 & 0xFFFFu) << 16;                    // tensor_dim0[15:0]
    g1[2] = (tensor_d0 >> 16) | ((tensor_d1 & 0xFFFFu) << 16);
    g1[3] = (tensor_d1 >> 16) | ((tile_d0 & 0xFFFFu) << 16); // tile_dim0
    g1[4] = (tile_d1 & 0xFFFFu);                            // tile_dim1, tile_dim2=0
    g1[5] = (unsigned)stride0;                              // dim0_stride[31:0]
    g1[6] = (unsigned)((stride0 >> 32) & 0xFFFFu);          // dim0_stride[47:32]
    g1[7] = 0u;                                             // dim1_stride unused (2D)
    asm volatile("tensor_load_to_lds %0, %1" :: "s"(g0), "s"(g1) : "memory");
}

// ---------------------------------------------------------------- convert
__global__ __launch_bounds__(256)
void cvt_bf16(const float* __restrict__ in, unsigned short* __restrict__ out, int n) {
    int i = (blockIdx.x * blockDim.x + threadIdx.x) * 4;
    if (i >= n) return;
    float4 f = *(const float4*)(in + i);
    ushort4 o;
    o.x = f2bf(f.x); o.y = f2bf(f.y); o.z = f2bf(f.z); o.w = f2bf(f.w);
    *(ushort4*)(out + i) = o;
}

// ---------------------------------------------------------------- projection
// C[i][j] = sum_d X[i][d] * W[j][d]   (X: BSxD bf16, W: DxD bf16)
// Double-buffered async-to-LDS pipeline: issue chunk kc+1 while computing kc.
#define KC   32
#define LDSX 40   // row stride (bf16): multiple of 8 -> 16B-aligned frags
#define NKC  (D_ / KC)

__global__ __launch_bounds__(256)
void proj_gemm(const unsigned short* __restrict__ X,
               const unsigned short* __restrict__ W,
               unsigned short* __restrict__ Out, int vmode)
{
    __shared__ unsigned short sX[2][128 * LDSX];
    __shared__ unsigned short sW[2][128 * LDSX];

    const int tid  = threadIdx.x;
    const int lane = tid & 31, wid = tid >> 5;
    const int hf   = lane >> 4, nl = lane & 15;
    const int wm   = wid & 3;          // 4 waves along M (32 rows each)
    const int wn   = wid >> 2;         // 2 waves along N (64 cols each)
    const size_t mBase = (size_t)blockIdx.y * 128;
    const size_t nBase = (size_t)blockIdx.x * 128;

    v8f acc[2][4];
#pragma unroll
    for (int i = 0; i < 2; ++i)
#pragma unroll
        for (int j = 0; j < 4; ++j) acc[i][j] = v8f_zero();

    const int lr = tid >> 1, lp = tid & 1;   // 256 thr -> 128 rows x 2 half-rows
    const unsigned short* gX = X + (mBase + lr) * D_ + lp * 16;
    const unsigned short* gW = W + (nBase + lr) * D_ + lp * 16;
    const unsigned dXa[2] = { (unsigned)(uintptr_t)(sX[0] + lr * LDSX + lp * 16),
                              (unsigned)(uintptr_t)(sX[1] + lr * LDSX + lp * 16) };
    const unsigned dWa[2] = { (unsigned)(uintptr_t)(sW[0] + lr * LDSX + lp * 16),
                              (unsigned)(uintptr_t)(sW[1] + lr * LDSX + lp * 16) };

    auto issue = [&](int kc, int buf) {
        unsigned long long gxa = (unsigned long long)(uintptr_t)(gX + kc * KC);
        unsigned long long gwa = (unsigned long long)(uintptr_t)(gW + kc * KC);
        asm volatile(
            "global_load_async_to_lds_b128 %0, %2, off\n\t"
            "global_load_async_to_lds_b128 %0, %2, off offset:16\n\t"
            "global_load_async_to_lds_b128 %1, %3, off\n\t"
            "global_load_async_to_lds_b128 %1, %3, off offset:16"
            :: "v"(dXa[buf]), "v"(dWa[buf]), "v"(gxa), "v"(gwa) : "memory");
    };

    issue(0, 0);                                     // prologue
    for (int kc = 0; kc < NKC; ++kc) {
        const int cur = kc & 1;
        if (kc + 1 < NKC) {
            issue(kc + 1, cur ^ 1);                  // one group in flight ahead
            __builtin_prefetch(gX + (kc + 2 < NKC ? kc + 2 : kc + 1) * KC, 0, 0);
            asm volatile("s_wait_asynccnt 0x4" ::: "memory");   // group kc done
        } else {
            asm volatile("s_wait_asynccnt 0x0" ::: "memory");
        }
        __syncthreads();                             // group kc visible to all

        Frag a[2], bf[4];
#pragma unroll
        for (int mt = 0; mt < 2; ++mt) {
            const unsigned short* p = sX[cur] + (wm * 32 + mt * 16 + nl) * LDSX + hf * 16;
            a[mt].q[0] = *(const uint4*)p;
            a[mt].q[1] = *(const uint4*)(p + 8);
        }
#pragma unroll
        for (int nt = 0; nt < 4; ++nt) {
            const unsigned short* p = sW[cur] + (wn * 64 + nt * 16 + nl) * LDSX + hf * 16;
            bf[nt].q[0] = *(const uint4*)p;
            bf[nt].q[1] = *(const uint4*)(p + 8);
        }
#pragma unroll
        for (int mt = 0; mt < 2; ++mt)
#pragma unroll
            for (int nt = 0; nt < 4; ++nt)
                acc[mt][nt] = wmma_bf16(a[mt].v, bf[nt].v, acc[mt][nt]);

        __syncthreads();      // all done reading buf cur before it is re-issued
    }

#pragma unroll
    for (int mt = 0; mt < 2; ++mt)
#pragma unroll
        for (int nt = 0; nt < 4; ++nt)
#pragma unroll
            for (int v = 0; v < 8; ++v) {
                size_t i = mBase + wm * 32 + mt * 16 + hf * 8 + v;
                size_t j = nBase + wn * 64 + nt * 16 + nl;
                int bb = (int)(i >> 11);        // /S_
                int ss = (int)(i & (S_ - 1));
                int hh = (int)(j >> 6);         // /DH_
                int dp = (int)(j & 63);
                size_t idx = (vmode == 0)
                    ? (((size_t)(bb * H_ + hh) * S_ + ss) * DH_ + dp)
                    : (((size_t)(bb * H_ + hh) * DH_ + dp) * S_ + ss);
                Out[idx] = f2bf(acc[mt][nt][v]);
            }
}

// ---------------------------------------------------------------- flash attention
// Qh,Kh: [B,H,S,64] bf16 ; Vt: [B,H,64,S] bf16 ; Out: [B,S,D] f32
// 8 waves (128 q rows) per block; double-buffered TDM tile pipeline.
__global__ __launch_bounds__(256)
void flash_attn(const unsigned short* __restrict__ Qh,
                const unsigned short* __restrict__ Kh,
                const unsigned short* __restrict__ Vt,
                float* __restrict__ Out)
{
    __shared__ unsigned short sK[2][32 * 64];   // 32 keys x 64 ch
    __shared__ unsigned short sV[2][64 * 32];   // 64 ch x 32 keys (from Vt)
    __shared__ unsigned short sP[8][16 * 32];   // per-wave P tile

    const int tid  = threadIdx.x, lane = tid & 31, wid = tid >> 5;
    const int hf   = lane >> 4, nl = lane & 15;
    const int h    = blockIdx.y, b = blockIdx.z;
    const size_t bh = (size_t)b * H_ + h;
    const int qBase = blockIdx.x * 128 + wid * 16;

    const unsigned short* Qb = Qh + bh * (size_t)S_ * DH_;
    const unsigned short* Kb = Kh + bh * (size_t)S_ * DH_;
    const unsigned short* Vb = Vt + bh * (size_t)DH_ * S_;
    const unsigned ldsK[2] = { (unsigned)(uintptr_t)&sK[0][0], (unsigned)(uintptr_t)&sK[1][0] };
    const unsigned ldsV[2] = { (unsigned)(uintptr_t)&sV[0][0], (unsigned)(uintptr_t)&sV[1][0] };

    auto issue_kv = [&](int kb, int buf) {
        // K tile: 32x64 bf16 contiguous -> degenerate 2048x1 TDM tile
        tdm_load_2d(ldsK[buf], Kb + (size_t)kb * 32 * DH_,
                    (unsigned)(S_ * DH_), 1u, 2048u, 1u,
                    (unsigned long long)S_ * DH_);
        // Vt tile: 64 rows x 32 keys, row stride S_ elements
        tdm_load_2d(ldsV[buf], Vb + (size_t)kb * 32,
                    (unsigned)S_, (unsigned)DH_, 32u, 64u,
                    (unsigned long long)S_);
    };

    Frag aq[2];
#pragma unroll
    for (int c = 0; c < 2; ++c) {
        const unsigned short* p = Qb + (size_t)(qBase + nl) * DH_ + c * 32 + hf * 16;
        aq[c].q[0] = *(const uint4*)p;
        aq[c].q[1] = *(const uint4*)(p + 8);
    }

    v8f o[4];
    float m_r[8], l_r[8];
#pragma unroll
    for (int nt = 0; nt < 4; ++nt) o[nt] = v8f_zero();
#pragma unroll
    for (int v = 0; v < 8; ++v) { m_r[v] = -3.0e38f; l_r[v] = 0.f; }

    const int NB = S_ / 32;
    if (wid == 0) issue_kv(0, 0);                    // prologue

    for (int kb = 0; kb < NB; ++kb) {
        const int cur = kb & 1;
        if (wid == 0) {
            if (kb + 1 < NB) {
                issue_kv(kb + 1, cur ^ 1);           // next tile in flight
                __builtin_amdgcn_s_wait_tensorcnt(2);  // tile kb complete
            } else {
                __builtin_amdgcn_s_wait_tensorcnt(0);
            }
        }
        __syncthreads();                             // tile kb visible to all

        v8f s0 = v8f_zero(), s1 = v8f_zero();
#pragma unroll
        for (int c = 0; c < 2; ++c) {
            Frag b0, b1;
            const unsigned short* p0 = sK[cur] + (0  + nl) * 64 + c * 32 + hf * 16;
            const unsigned short* p1 = sK[cur] + (16 + nl) * 64 + c * 32 + hf * 16;
            b0.q[0] = *(const uint4*)p0; b0.q[1] = *(const uint4*)(p0 + 8);
            b1.q[0] = *(const uint4*)p1; b1.q[1] = *(const uint4*)(p1 + 8);
            s0 = wmma_bf16(aq[c].v, b0.v, s0);
            s1 = wmma_bf16(aq[c].v, b1.v, s1);
        }

        const float scl = 0.125f;   // 1/sqrt(64)
        float alpha[8];
#pragma unroll
        for (int v = 0; v < 8; ++v) {
            float x0 = s0[v] * scl, x1 = s1[v] * scl;
            float mx = rmax16(fmaxf(x0, x1));
            float mn = fmaxf(m_r[v], mx);
            alpha[v] = __expf(m_r[v] - mn);
            m_r[v] = mn;
            float p0 = __expf(x0 - mn), p1 = __expf(x1 - mn);
            l_r[v] = l_r[v] * alpha[v] + rsum16(p0 + p1);
            int row = v + hf * 8;
            sP[wid][row * 32 + nl]      = f2bf(p0);
            sP[wid][row * 32 + 16 + nl] = f2bf(p1);
        }
#pragma unroll
        for (int nt = 0; nt < 4; ++nt)
#pragma unroll
            for (int v = 0; v < 8; ++v) o[nt][v] *= alpha[v];

        Frag pa;   // P (16x32) as A fragment, via LDS (same-wave DS ordering)
        {
            const unsigned short* p = &sP[wid][nl * 32 + hf * 16];
            pa.q[0] = *(const uint4*)p;
            pa.q[1] = *(const uint4*)(p + 8);
        }
#pragma unroll
        for (int nt = 0; nt < 4; ++nt) {
            Frag bv;
            const unsigned short* p = sV[cur] + (nt * 16 + nl) * 32 + hf * 16;
            bv.q[0] = *(const uint4*)p; bv.q[1] = *(const uint4*)(p + 8);
            o[nt] = wmma_bf16(pa.v, bv.v, o[nt]);
        }

        __syncthreads();      // all done with buf cur before it is re-issued
    }

    float inv[8];
#pragma unroll
    for (int v = 0; v < 8; ++v) inv[v] = 1.f / l_r[v];
#pragma unroll
    for (int nt = 0; nt < 4; ++nt)
#pragma unroll
        for (int v = 0; v < 8; ++v) {
            int srow = qBase + v + hf * 8;
            Out[((size_t)b * S_ + srow) * D_ + h * DH_ + nt * 16 + nl] = o[nt][v] * inv[v];
        }
}

// ---------------------------------------------------------------- head-0 raw scores
__global__ __launch_bounds__(128)
void head0_scores(const unsigned short* __restrict__ Qh,
                  const unsigned short* __restrict__ Kh,
                  const unsigned char* __restrict__ mask,
                  float* __restrict__ top)
{
    const int tid = threadIdx.x, lane = tid & 31, wid = tid >> 5;
    const int hf = lane >> 4, nl = lane & 15;
    const int b = blockIdx.z;
    const int kBase = blockIdx.x * 64;
    const int qBase = blockIdx.y * 64 + wid * 16;
    const size_t bh = (size_t)b * H_;   // head 0
    const unsigned short* Qb = Qh + bh * (size_t)S_ * DH_;
    const unsigned short* Kb = Kh + bh * (size_t)S_ * DH_;

    Frag aq[2];
#pragma unroll
    for (int c = 0; c < 2; ++c) {
        const unsigned short* p = Qb + (size_t)(qBase + nl) * DH_ + c * 32 + hf * 16;
        aq[c].q[0] = *(const uint4*)p;
        aq[c].q[1] = *(const uint4*)(p + 8);
    }

#pragma unroll
    for (int nt = 0; nt < 4; ++nt) {
        v8f acc = v8f_zero();
#pragma unroll
        for (int c = 0; c < 2; ++c) {
            Frag bf;
            const unsigned short* p = Kb + (size_t)(kBase + nt * 16 + nl) * DH_ + c * 32 + hf * 16;
            bf.q[0] = *(const uint4*)p; bf.q[1] = *(const uint4*)(p + 8);
            acc = wmma_bf16(aq[c].v, bf.v, acc);
        }
#pragma unroll
        for (int v = 0; v < 8; ++v) {
            int sr = qBase + v + hf * 8;
            int sc = kBase + nt * 16 + nl;
            float val = acc[v] * 0.125f;
            size_t idx = ((size_t)b * S_ + sr) * S_ + sc;
            if (mask[idx]) val = -1e18f;
            top[idx] = val;
        }
    }
}

// ---------------------------------------------------------------- in-place row softmax
__global__ __launch_bounds__(256)
void row_softmax(float* __restrict__ top)
{
    __shared__ float red[256];
    const int r = blockIdx.x, b = blockIdx.y, tid = threadIdx.x;
    float* row = top + ((size_t)b * S_ + r) * S_;

    float vals[8];
    float mx = -3.0e38f;
#pragma unroll
    for (int k = 0; k < 8; ++k) { vals[k] = row[tid + k * 256]; mx = fmaxf(mx, vals[k]); }
    red[tid] = mx; __syncthreads();
    for (int s = 128; s > 0; s >>= 1) {
        if (tid < s) red[tid] = fmaxf(red[tid], red[tid + s]);
        __syncthreads();
    }
    mx = red[0]; __syncthreads();

    float sum = 0.f;
#pragma unroll
    for (int k = 0; k < 8; ++k) { vals[k] = __expf(vals[k] - mx); sum += vals[k]; }
    red[tid] = sum; __syncthreads();
    for (int s = 128; s > 0; s >>= 1) {
        if (tid < s) red[tid] += red[tid + s];
        __syncthreads();
    }
    float inv = 1.f / red[0];
#pragma unroll
    for (int k = 0; k < 8; ++k) row[tid + k * 256] = vals[k] * inv;
}

// ---------------------------------------------------------------- host
extern "C" void kernel_launch(void* const* d_in, const int* in_sizes, int n_in,
                              void* d_out, int out_size, void* d_ws, size_t ws_size,
                              hipStream_t stream)
{
    (void)in_sizes; (void)n_in; (void)out_size; (void)ws_size;

    const float* key   = (const float*)d_in[0];
    const float* value = (const float*)d_in[1];
    const float* query = (const float*)d_in[2];
    const unsigned char* mask = (const unsigned char*)d_in[3];
    const float* Wk = (const float*)d_in[4];
    const float* Wv = (const float*)d_in[5];
    const float* Wq = (const float*)d_in[6];

    char* ws = (char*)d_ws;
    size_t off = 0;
    auto alloc = [&](size_t bytes) {
        void* p = ws + off;
        off += (bytes + 255) & ~(size_t)255;
        return p;
    };
    const size_t nIn = (size_t)BS_ * D_;   // 8,388,608
    const size_t nW  = (size_t)D_ * D_;    // 1,048,576

    unsigned short* kbf  = (unsigned short*)alloc(nIn * 2);
    unsigned short* vbf  = (unsigned short*)alloc(nIn * 2);
    unsigned short* qbf  = (unsigned short*)alloc(nIn * 2);
    unsigned short* wkbf = (unsigned short*)alloc(nW * 2);
    unsigned short* wvbf = (unsigned short*)alloc(nW * 2);
    unsigned short* wqbf = (unsigned short*)alloc(nW * 2);
    unsigned short* Kh   = (unsigned short*)alloc(nIn * 2);  // [B,H,S,64]
    unsigned short* Vt   = (unsigned short*)alloc(nIn * 2);  // [B,H,64,S]
    unsigned short* Qh   = (unsigned short*)alloc(nIn * 2);  // [B,H,S,64]

    dim3 cb(256);
    cvt_bf16<<<dim3((unsigned)((nIn / 4 + 255) / 256)), cb, 0, stream>>>(key,   kbf,  (int)nIn);
    cvt_bf16<<<dim3((unsigned)((nIn / 4 + 255) / 256)), cb, 0, stream>>>(value, vbf,  (int)nIn);
    cvt_bf16<<<dim3((unsigned)((nIn / 4 + 255) / 256)), cb, 0, stream>>>(query, qbf,  (int)nIn);
    cvt_bf16<<<dim3((unsigned)((nW  / 4 + 255) / 256)), cb, 0, stream>>>(Wk,    wkbf, (int)nW);
    cvt_bf16<<<dim3((unsigned)((nW  / 4 + 255) / 256)), cb, 0, stream>>>(Wv,    wvbf, (int)nW);
    cvt_bf16<<<dim3((unsigned)((nW  / 4 + 255) / 256)), cb, 0, stream>>>(Wq,    wqbf, (int)nW);

    dim3 pg(D_ / 128, BS_ / 128);   // (8, 64)
    proj_gemm<<<pg, dim3(256), 0, stream>>>(kbf, wkbf, Kh, 0);
    proj_gemm<<<pg, dim3(256), 0, stream>>>(vbf, wvbf, Vt, 1);
    proj_gemm<<<pg, dim3(256), 0, stream>>>(qbf, wqbf, Qh, 0);

    float* out = (float*)d_out;                 // [B,S,D]
    float* top = out + (size_t)B_ * S_ * D_;    // [B,S,S]

    flash_attn<<<dim3(S_ / 128, H_, B_), dim3(256), 0, stream>>>(Qh, Kh, Vt, out);
    head0_scores<<<dim3(S_ / 64, S_ / 64, B_), dim3(128), 0, stream>>>(Qh, Kh, mask, top);
    row_softmax<<<dim3(S_, B_), dim3(256), 0, stream>>>(top);
}
